// CustomRNN_31104153157802
// MI455X (gfx1250) — matrix operand order repaired and checked by reading, not compiled, not run on hardware
//
#include <hip/hip_runtime.h>

// ---------------------------------------------------------------------------
// RNN scan on MI455X (gfx1250): persistent kernel, bf16 WMMA, f32 accumulate,
// 2x2 register blocking (32x32 output block per wave => 4 wmma per K-step,
// halving L2 fragment traffic vs 1 tile/wave).
//   h_{t+1} = h_t @ W2^T + x[:,t] (x) w1      (8x32 = 256 blocks)
//   y_t     = h_{t+1} @ W3^T                  (8 blocks)
// 264 wave32s = 33 WGs x 256 threads; grid barrier between steps.
// ---------------------------------------------------------------------------

typedef __bf16 bf16_t;
typedef bf16_t v16bf __attribute__((ext_vector_type(16)));
typedef float  v8f   __attribute__((ext_vector_type(8)));

#define B_      256
#define T_      4096
#define H_      1024
#define O_      32
#define NWG     33
#define THREADS 256

#define WMMA_BF16(a, b, c) __builtin_amdgcn_wmma_f32_16x16x32_bf16( \
    false, (a), false, (b), (short)0, (c), false, false)

// f32 -> bf16, round-to-nearest-even
__device__ __forceinline__ unsigned short f2bf(float f) {
    union { float f; unsigned u; } cv; cv.f = f;
    unsigned u = cv.u;
    return (unsigned short)((u + 0x7fffu + ((u >> 16) & 1u)) >> 16);
}

// ---------------------------------------------------------------------------
// Pack a row-major f32 weight W (rows = N outputs, row length H_) into the
// WMMA B-fragment bf16 layout:
//   fragment(nt, ks): lane l holds B[k, n] with n = nt*16 + (l&15),
//                     k = ks*32 + 16*(j>>3) + 8*(l>>4) + (j&7),  j = 0..15
// Flat index: ((nt*32 + ks)*32 + lane)*16 + j
// ---------------------------------------------------------------------------
__global__ void pack_w_kernel(const float* __restrict__ W,
                              unsigned short* __restrict__ wp, int total) {
    int i = blockIdx.x * blockDim.x + threadIdx.x;
    if (i >= total) return;
    int j    = i & 15;
    int lane = (i >> 4) & 31;
    int ks   = (i >> 9) & 31;
    int nt   = i >> 14;
    int n = nt * 16 + (lane & 15);
    int k = ks * 32 + ((j >> 3) << 4) + ((lane >> 4) << 3) + (j & 7);
    wp[i] = f2bf(W[n * H_ + k]);
}

// Zero both h fragment buffers (h0 = 0) and the barrier counter.
__global__ void zero_ws_kernel(unsigned* __restrict__ hAB,
                               unsigned* __restrict__ counter, int total_u32) {
    int i = blockIdx.x * blockDim.x + threadIdx.x;
    if (i == 0) *counter = 0u;
    if (i < total_u32) hAB[i] = 0u;
}

// ---------------------------------------------------------------------------
// Persistent RNN kernel. 33 WGs x 256 threads = 264 waves.
//   wave 0..255 : h block  (mt2 = w/32, nt2 = w%32), 32x32 output, 4 wmma/K
//   wave 256+   : y block  (mt2 = w-256), 32 rows x all 32 outputs, computed
//                 for t-1 from the buffer finished last step (double buffer)
// ---------------------------------------------------------------------------
__global__ __launch_bounds__(THREADS, 1)
void rnn_persistent(const float* __restrict__ x, const float* __restrict__ W1,
                    const unsigned short* __restrict__ w2p,
                    const unsigned short* __restrict__ w3p,
                    unsigned short* __restrict__ hA,
                    unsigned short* __restrict__ hB,
                    float* __restrict__ out,
                    unsigned* __restrict__ counter) {
    const int tid  = threadIdx.x;
    const int lane = tid & 31;
    const int wid  = ((int)blockIdx.x * THREADS + tid) >> 5;
    const int hi   = lane >> 4;   // half-wave id
    const int nl   = lane & 15;

    const bool is_h = (wid < 256);
    int mt2 = 0, nt2 = 0;
    if (is_h) { mt2 = wid >> 5; nt2 = wid & 31; }
    else      { mt2 = wid - 256; }

    // Producer-side D -> A-fragment mapping (h blocks only).
    // Even N-tile (n = nt2*32 + nl):      k_local = nl      -> j = nl&7
    // Odd  N-tile (n = nt2*32 + 16 + nl): k_local = 16 + nl -> j = 8 + (nl&7)
    // Both share ks2 = nt2 and hik = nl>>3.
    const int hik    = nl >> 3;
    const int j_even = nl & 7;
    const int j_odd  = 8 + (nl & 7);
    float w1a = 0.f, w1b = 0.f;
    if (is_h) {
        w1a = W1[nt2 * 32 + nl];
        w1b = W1[nt2 * 32 + 16 + nl];
    }

    for (int t = 0; t <= T_; ++t) {
        const unsigned short* cur = (t & 1) ? hB : hA;
        unsigned short*       nxt = (t & 1) ? hA : hB;

        if (is_h) {
            if (t < T_) {
                // A tiles: rows [mt2*32, +32) => fragment tiles 2*mt2, 2*mt2+1
                // B tiles: cols [nt2*32, +32) => fragment tiles 2*nt2, 2*nt2+1
                const v16bf* ap = (const v16bf*)cur + (mt2 * 2 * 32) * 32 + lane;
                const v16bf* bp = (const v16bf*)w2p + (nt2 * 2 * 32) * 32 + lane;
                v8f a00 = {}, a01 = {}, a10 = {}, a11 = {};
#pragma unroll 4
                for (int ks = 0; ks < 32; ++ks) {
                    v16bf fA0 = ap[ks * 32];
                    v16bf fA1 = ap[ks * 32 + 1024];
                    v16bf fB0 = bp[ks * 32];
                    v16bf fB1 = bp[ks * 32 + 1024];
                    a00 = WMMA_BF16(fA0, fB0, a00);
                    a01 = WMMA_BF16(fA0, fB1, a01);
                    a10 = WMMA_BF16(fA1, fB0, a10);
                    a11 = WMMA_BF16(fA1, fB1, a11);
                }
                // h_new += x[:,t] (x) w1 in f32, round to bf16 A-fragments
                const int base0 = ((2 * mt2) * 32 + nt2) * 512 + (hik << 8);
                const int base1 = ((2 * mt2 + 1) * 32 + nt2) * 512 + (hik << 8);
#pragma unroll
                for (int v = 0; v < 8; ++v) {
                    int mlocal = v + (hi << 3);
                    float xa = x[(mt2 * 32 + mlocal) * T_ + t];
                    float xb = x[(mt2 * 32 + 16 + mlocal) * T_ + t];
                    nxt[base0 + (mlocal << 4) + j_even] = f2bf(a00[v] + xa * w1a);
                    nxt[base0 + (mlocal << 4) + j_odd]  = f2bf(a01[v] + xa * w1b);
                    nxt[base1 + (mlocal << 4) + j_even] = f2bf(a10[v] + xb * w1a);
                    nxt[base1 + (mlocal << 4) + j_odd]  = f2bf(a11[v] + xb * w1b);
                }
            }
        } else {
            if (t > 0) {  // y_{t-1}: rows [mt2*32,+32) x all O=32 outputs
                const v16bf* ap = (const v16bf*)cur + (mt2 * 2 * 32) * 32 + lane;
                const v16bf* bp = (const v16bf*)w3p + lane;
                v8f a00 = {}, a01 = {}, a10 = {}, a11 = {};
#pragma unroll 4
                for (int ks = 0; ks < 32; ++ks) {
                    v16bf fA0 = ap[ks * 32];
                    v16bf fA1 = ap[ks * 32 + 1024];
                    v16bf fB0 = bp[ks * 32];
                    v16bf fB1 = bp[ks * 32 + 1024];
                    a00 = WMMA_BF16(fA0, fB0, a00);
                    a01 = WMMA_BF16(fA0, fB1, a01);
                    a10 = WMMA_BF16(fA1, fB0, a10);
                    a11 = WMMA_BF16(fA1, fB1, a11);
                }
                const int tp = t - 1;
#pragma unroll
                for (int v = 0; v < 8; ++v) {
                    int b0 = mt2 * 32 + v + (hi << 3);
                    int b1 = b0 + 16;
                    out[(b0 * T_ + tp) * O_ + nl]      = a00[v];
                    out[(b0 * T_ + tp) * O_ + 16 + nl] = a01[v];
                    out[(b1 * T_ + tp) * O_ + nl]      = a10[v];
                    out[(b1 * T_ + tp) * O_ + 16 + nl] = a11[v];
                }
            }
        }

        if (t < T_) {  // device-wide barrier between steps
            __syncthreads();
            if (tid == 0) {
                __threadfence();                    // release h_new writes
                atomicAdd(counter, 1u);
                const unsigned target = (unsigned)NWG * (unsigned)(t + 1);
                while (atomicAdd(counter, 0u) < target)
                    __builtin_amdgcn_s_sleep(1);
            }
            __syncthreads();
            __threadfence();                        // acquire other WGs' writes
        }
    }
}

extern "C" void kernel_launch(void* const* d_in, const int* in_sizes, int n_in,
                              void* d_out, int out_size, void* d_ws, size_t ws_size,
                              hipStream_t stream) {
    const float* x  = (const float*)d_in[0];   // (B, T)
    const float* W1 = (const float*)d_in[1];   // (H, 1)
    const float* W2 = (const float*)d_in[2];   // (H, H)
    const float* W3 = (const float*)d_in[3];   // (O, H)
    float* out = (float*)d_out;                // (B, T, O) f32

    // Workspace layout (bytes):
    //   [0,256)        barrier counter
    //   [256, +512K)   hA  (B x H bf16, A-fragment packed)
    //   [.., +512K)    hB
    //   [.., +2M)      W2 packed (B-fragment bf16)
    //   [.., +64K)     W3 packed
    char* ws = (char*)d_ws;
    unsigned*       counter = (unsigned*)ws;
    unsigned short* hA  = (unsigned short*)(ws + 256);
    unsigned short* hB  = (unsigned short*)(ws + 256 + 524288);
    unsigned short* w2p = (unsigned short*)(ws + 256 + 2 * 524288);
    unsigned short* w3p = (unsigned short*)(ws + 256 + 2 * 524288 + 2097152);

    const int h_u32 = (2 * B_ * H_ * 2) / 4;   // both h buffers, as u32
    zero_ws_kernel<<<(h_u32 + 255) / 256, 256, 0, stream>>>((unsigned*)hA, counter, h_u32);
    pack_w_kernel<<<(H_ * H_ + 255) / 256, 256, 0, stream>>>(W2, w2p, H_ * H_);
    pack_w_kernel<<<(O_ * H_ + 255) / 256, 256, 0, stream>>>(W3, w3p, O_ * H_);
    rnn_persistent<<<NWG, THREADS, 0, stream>>>(x, W1, w2p, w3p, hA, hB, out, counter);
}
